// DecoderLayer_36215164239989
// MI455X (gfx1250) — matrix operand order repaired and checked by reading, not compiled, not run on hardware
//
#include <hip/hip_runtime.h>

typedef __attribute__((ext_vector_type(16))) __bf16 v16bf;
typedef __attribute__((ext_vector_type(8)))  float  v8f;
typedef __attribute__((ext_vector_type(4)))  unsigned v4u;
typedef __attribute__((ext_vector_type(8)))  int  v8i;
typedef __attribute__((ext_vector_type(4)))  int  v4i;

#define DEVFN __device__ __forceinline__

DEVFN unsigned short f2bf(float f) {
  unsigned u = __float_as_uint(f);
  u += 0x7fffu + ((u >> 16) & 1u);       // round-to-nearest-even
  return (unsigned short)(u >> 16);
}

union ABf { v16bf v; unsigned u[8]; };

DEVFN v8f wmma_bf16(const ABf& a, const ABf& b, v8f c) {
  return __builtin_amdgcn_wmma_f32_16x16x32_bf16(
      /*neg_a=*/false, a.v, /*neg_b=*/false, b.v,
      /*c_mod=*/(short)0, c, /*reuse_a=*/false, /*reuse_b=*/false);
}

// ------------------------------------------------------------------ TDM load
// 2D tile load Global->LDS via Tensor Data Mover (D# per CDNA5 ISA).
// Tile: 32 elements wide (64B rows) x `rows` high, element = bf16 (2B).
// LDS rows padded to 40 elements (pad_interval=16 DW, pad_amount=4 DW).
DEVFN void tdm_load_tile(unsigned lds_bytes, const void* gptr,
                         unsigned stride_elems, unsigned rows) {
  unsigned long long ga = (unsigned long long)(size_t)gptr;
  v4u g0;
  g0[0] = 1u;                                   // count=1 valid, user mode
  g0[1] = lds_bytes;                            // lds_addr
  g0[2] = (unsigned)ga;                         // global_addr[31:0]
  g0[3] = (unsigned)(ga >> 32) | (2u << 30);    // global_addr[56:32] | type=2
  v8i g1;
  g1[0] = (int)((1u << 16)                      // data_size = 2 bytes
              | (1u << 20)                      // pad_enable
              | (3u << 22)                      // pad_interval: 16 DWORDs (64B)
              | (3u << 25));                    // pad_amount: 4 DWORDs (16B)
  g1[1] = (int)((stride_elems & 0xFFFFu) << 16);                 // tensor_dim0[15:0]
  g1[2] = (int)((stride_elems >> 16) | ((rows & 0xFFFFu) << 16));// dim0[31:16]|dim1[15:0]
  g1[3] = (int)((rows >> 16) | (32u << 16));                     // dim1[31:16]|tile_dim0=32
  g1[4] = (int)(rows & 0xFFFFu);                                 // tile_dim1=rows, tile_dim2=0
  g1[5] = (int)stride_elems;                                     // tensor_dim0_stride[31:0]
  g1[6] = 0;                                                     // stride hi / dim1_stride
  g1[7] = 0;
  v4i z4 = {0, 0, 0, 0};
  v8i z8 = {0, 0, 0, 0, 0, 0, 0, 0};
  // 6-arg form (this toolchain): (v4u, v8i, v4i, v4i, v8i, i32 cpol)
  __builtin_amdgcn_tensor_load_to_lds(g0, g1, z4, z4, z8, 0);
}

// ---------------------------------------------------------------- f32 -> bf16
__global__ void k_cvt(const float* __restrict__ in, unsigned short* __restrict__ out, int n) {
  int i = blockIdx.x * blockDim.x + threadIdx.x;
  if (i < n) out[i] = f2bf(in[i]);
}

// -------------------------------------------- weight transpose + f32 -> bf16
// in: [K][N] f32 row-major  ->  out: [N][K] bf16 row-major. 32x32 LDS tiles.
__global__ __launch_bounds__(256) void k_cvt_t(const float* __restrict__ in,
                                               unsigned short* __restrict__ out,
                                               int K, int N) {
  __shared__ unsigned short t[32][33];
  const int n0 = blockIdx.x * 32, k0 = blockIdx.y * 32;
  const int tx = threadIdx.x & 31, ty = threadIdx.x >> 5;   // 32 x 8
  #pragma unroll
  for (int i = 0; i < 4; ++i) {
    int k = ty + i * 8;
    t[tx][k] = f2bf(in[(size_t)(k0 + k) * N + n0 + tx]);    // t[n][k]
  }
  __syncthreads();
  #pragma unroll
  for (int i = 0; i < 4; ++i) {
    int n = ty + i * 8;
    out[(size_t)(n0 + n) * K + k0 + tx] = t[n][tx];
  }
}

// ---------------------------------------------------------------- bf16 GEMM
// C[M,N] = A[M,K] * Bt[N,K]^T + bias (+resid) (ReLU), fp32 accum via WMMA.
// 256 threads = 8 waves; tile 128x128x32; wave tile 32x64 (2x4 WMMA).
// A and Bt tiles DMA'd into double-buffered LDS by the Tensor Data Mover.
template<bool RELU, bool RESID, bool WF32, bool WB16>
__global__ __launch_bounds__(256) void k_gemm(
    const unsigned short* __restrict__ A,   // [M][K]
    const unsigned short* __restrict__ Bt,  // [N][K] (pre-transposed weights)
    const float* __restrict__ bias, const float* __restrict__ resid,
    float* __restrict__ Cf, unsigned short* __restrict__ Cb,
    int M, int N, int K)
{
  constexpr int BM = 128, BN = 128, BK = 32, BKP = 40;  // 80B LDS row stride
  __shared__ unsigned short As[2][BM * BKP];
  __shared__ unsigned short Bs[2][BN * BKP];
  const int tid = threadIdx.x;
  const int m0 = blockIdx.y * BM, n0 = blockIdx.x * BN;
  const int w = tid >> 5, lane = tid & 31, half = lane >> 4, lr = lane & 15;
  const int wm = (w & 3) * 32, wn = (w >> 2) * 64;

  v8f zero = {0.f,0.f,0.f,0.f,0.f,0.f,0.f,0.f};
  v8f acc[2][4];
  #pragma unroll
  for (int mt = 0; mt < 2; ++mt)
    #pragma unroll
    for (int nt = 0; nt < 4; ++nt) acc[mt][nt] = zero;

  auto issue = [&](int buf, int k0) {
    tdm_load_tile((unsigned)(size_t)(void*)&As[buf][0],
                  A + (size_t)m0 * K + k0, (unsigned)K, (unsigned)BM);
    tdm_load_tile((unsigned)(size_t)(void*)&Bs[buf][0],
                  Bt + (size_t)n0 * K + k0, (unsigned)K, (unsigned)BN);
  };

  if (tid < 32) issue(0, 0);                     // EXEC-independent, wave 0 only
  const int nk = K / BK;
  for (int ki = 0; ki < nk; ++ki) {
    const int cur = ki & 1;
    if (tid < 32) __builtin_amdgcn_s_wait_tensorcnt(0);
    __syncthreads();                             // tile `cur` visible to all waves
    if ((ki + 1 < nk) && tid < 32) issue(cur ^ 1, (ki + 1) * BK);

    const unsigned short* Asc = &As[cur][0];
    const unsigned short* Bsc = &Bs[cur][0];
    ABf af[2], bf[4];
    #pragma unroll
    for (int mt = 0; mt < 2; ++mt) {
      int row = wm + mt * 16 + lr;
      #pragma unroll
      for (int i = 0; i < 8; ++i) {              // 16-bit A 16x32 layout
        int kk = (i < 4) ? (half * 8 + 2 * i) : (16 + half * 8 + 2 * (i - 4));
        af[mt].u[i] = *(const unsigned*)(&Asc[row * BKP + kk]);
      }
    }
    #pragma unroll
    for (int nt = 0; nt < 4; ++nt) {
      int col = wn + nt * 16 + lr;
      #pragma unroll
      for (int i = 0; i < 8; ++i) {              // 16-bit B 32x16 layout
        int kk = (i < 4) ? (half * 8 + 2 * i) : (16 + half * 8 + 2 * (i - 4));
        bf[nt].u[i] = *(const unsigned*)(&Bsc[col * BKP + kk]);
      }
    }
    #pragma unroll
    for (int mt = 0; mt < 2; ++mt)
      #pragma unroll
      for (int nt = 0; nt < 4; ++nt)
        acc[mt][nt] = wmma_bf16(af[mt], bf[nt], acc[mt][nt]);
    __syncthreads();                             // done reading tile `cur`
  }

  // Epilogue: C layout = lane col, VGPR r -> row r + 8*half
  #pragma unroll
  for (int mt = 0; mt < 2; ++mt)
    #pragma unroll
    for (int nt = 0; nt < 4; ++nt) {
      int n = n0 + wn + nt * 16 + lr;
      float bv = bias[n];
      #pragma unroll
      for (int r = 0; r < 8; ++r) {
        int mI = m0 + wm + mt * 16 + r + 8 * half;
        float val = acc[mt][nt][r] + bv;
        if (RESID) val += resid[(size_t)mI * N + n];
        if (RELU)  val = fmaxf(val, 0.f);
        if (WF32)  Cf[(size_t)mI * N + n] = val;
        if (WB16)  Cb[(size_t)mI * N + n] = f2bf(val);
      }
    }
}

// ---------------------------------------------------------------- QKV repack
// [B*T, 3*D] -> [3, B, H, T, 64] (head-major so attention loads are contiguous)
__global__ void k_repack(const unsigned short* __restrict__ qkv, unsigned short* __restrict__ out) {
  int i = blockIdx.x * 256 + threadIdx.x;           // over 8192*3072, exact grid
  int n = i % 3072, row = i / 3072;
  int which = n >> 10, h = (n >> 6) & 15, d = n & 63;
  int b = row >> 11, t = row & 2047;
  size_t dst = (size_t)which * (4 * 16 * 2048 * 64)
             + (((size_t)(b * 16 + h)) * 2048 + t) * 64 + d;
  out[dst] = qkv[i];
}

// ---------------------------------------------------------------- flash attention
// One wave32 per (b, h, 16-row q tile); stream 32 kv per step; online softmax.
__global__ __launch_bounds__(32) void k_attn(
    const unsigned short* __restrict__ Qh, const unsigned short* __restrict__ Kh,
    const unsigned short* __restrict__ Vh, unsigned short* __restrict__ Oa)
{
  const int T = 2048, HD = 64, H = 16, D = 1024;
  const int qt = blockIdx.x, h = blockIdx.y, b = blockIdx.z;
  const int qbase = qt * 16;
  const int lane = threadIdx.x, half = lane >> 4, lr = lane & 15;
  const size_t hoff = ((size_t)(b * H + h)) * T * HD;
  const unsigned short* q  = Qh + hoff;
  const unsigned short* kp = Kh + hoff;
  const unsigned short* vp = Vh + hoff;

  __shared__ unsigned short pls[16 * 34];           // P bounce: C-layout -> A-layout

  ABf qf[2];                                        // Q tile 16x64 = 2 A frags
  #pragma unroll
  for (int kt = 0; kt < 2; ++kt)
    #pragma unroll
    for (int i = 0; i < 8; ++i) {
      int kk = kt * 32 + ((i < 4) ? (half * 8 + 2 * i) : (16 + half * 8 + 2 * (i - 4)));
      qf[kt].u[i] = *(const unsigned*)(q + (size_t)(qbase + lr) * HD + kk);
    }

  v8f zero = {0.f,0.f,0.f,0.f,0.f,0.f,0.f,0.f};
  v8f o[4];
  float m[8], l[8];
  #pragma unroll
  for (int ot = 0; ot < 4; ++ot) o[ot] = zero;
  #pragma unroll
  for (int r = 0; r < 8; ++r) { m[r] = -3.0e38f; l[r] = 0.f; }

  const int nblk = (qbase + 15) / 32 + 1;
  for (int kb = 0; kb < nblk; ++kb) {
    const int kv0 = kb * 32;
    v8f s[2]; s[0] = zero; s[1] = zero;
    #pragma unroll
    for (int nt = 0; nt < 2; ++nt) {
      int kvp = kv0 + nt * 16 + lr;                 // B-frag col = kv position
      #pragma unroll
      for (int kt = 0; kt < 2; ++kt) {
        ABf kf;                                     // K^T frag: hd is contiguous
        #pragma unroll
        for (int i = 0; i < 8; ++i) {
          int kk = kt * 32 + ((i < 4) ? (half * 8 + 2 * i) : (16 + half * 8 + 2 * (i - 4)));
          kf.u[i] = *(const unsigned*)(kp + (size_t)kvp * HD + kk);
        }
        s[nt] = wmma_bf16(qf[kt], kf, s[nt]);
      }
    }
    float sv[2][8], mb[8];
    #pragma unroll
    for (int r = 0; r < 8; ++r) mb[r] = -3.0e38f;
    #pragma unroll
    for (int nt = 0; nt < 2; ++nt)
      #pragma unroll
      for (int r = 0; r < 8; ++r) {
        int kvp = kv0 + nt * 16 + lr;
        int qp  = qbase + r + 8 * half;
        float x = s[nt][r] * 0.125f;                // 1/sqrt(64)
        if (kvp > qp) x = -3.0e38f;
        sv[nt][r] = x;
        mb[r] = fmaxf(mb[r], x);
      }
    #pragma unroll
    for (int dlt = 1; dlt < 16; dlt <<= 1)
      #pragma unroll
      for (int r = 0; r < 8; ++r)
        mb[r] = fmaxf(mb[r], __shfl_xor(mb[r], dlt, 16));
    float ef[8], ps[8];
    #pragma unroll
    for (int r = 0; r < 8; ++r) {
      float mn = fmaxf(m[r], mb[r]);
      ef[r] = __expf(m[r] - mn);
      m[r] = mn;
      ps[r] = 0.f;
    }
    #pragma unroll
    for (int nt = 0; nt < 2; ++nt)
      #pragma unroll
      for (int r = 0; r < 8; ++r) {
        float pv = __expf(sv[nt][r] - m[r]);
        sv[nt][r] = pv;
        ps[r] += pv;
      }
    #pragma unroll
    for (int dlt = 1; dlt < 16; dlt <<= 1)
      #pragma unroll
      for (int r = 0; r < 8; ++r)
        ps[r] += __shfl_xor(ps[r], dlt, 16);
    #pragma unroll
    for (int r = 0; r < 8; ++r) l[r] = l[r] * ef[r] + ps[r];
    #pragma unroll
    for (int ot = 0; ot < 4; ++ot)
      #pragma unroll
      for (int r = 0; r < 8; ++r) o[ot][r] *= ef[r];

    // P: C-layout -> LDS -> A-layout frag
    #pragma unroll
    for (int nt = 0; nt < 2; ++nt)
      #pragma unroll
      for (int r = 0; r < 8; ++r)
        pls[(r + 8 * half) * 34 + nt * 16 + lr] = f2bf(sv[nt][r]);
    __syncthreads();
    ABf pf;
    #pragma unroll
    for (int i = 0; i < 8; ++i) {
      int kk = (i < 4) ? (half * 8 + 2 * i) : (16 + half * 8 + 2 * (i - 4));
      pf.u[i] = *(const unsigned*)(&pls[lr * 34 + kk]);
    }
    __syncthreads();

    // O += P(16x32) * V(32x64): V B-frag needs (kv,kv+1) pairs -> 2 b16 gathers
    #pragma unroll
    for (int ot = 0; ot < 4; ++ot) {
      ABf vf;
      int col = ot * 16 + lr;
      #pragma unroll
      for (int i = 0; i < 8; ++i) {
        int kk = (i < 4) ? (half * 8 + 2 * i) : (16 + half * 8 + 2 * (i - 4));
        unsigned lo = vp[(size_t)(kv0 + kk) * HD + col];
        unsigned hi = vp[(size_t)(kv0 + kk + 1) * HD + col];
        vf.u[i] = lo | (hi << 16);
      }
      o[ot] = wmma_bf16(pf, vf, o[ot]);
    }
  }
  #pragma unroll
  for (int ot = 0; ot < 4; ++ot)
    #pragma unroll
    for (int r = 0; r < 8; ++r) {
      int qp = qbase + r + 8 * half;
      float val = o[ot][r] / l[r];
      Oa[((size_t)(b * T + qp)) * D + h * HD + ot * 16 + lr] = f2bf(val);
    }
}

// ---------------------------------------------------------------- LayerNorm
template<bool WB16>
__global__ __launch_bounds__(256) void k_ln(const float* __restrict__ X,
    const float* __restrict__ gam, const float* __restrict__ bet,
    float* __restrict__ Yf, unsigned short* __restrict__ Yb)
{
  const int D = 1024;
  __shared__ float red[256];
  const int row = blockIdx.x, t = threadIdx.x;
  const float* xr = X + (size_t)row * D;
  float v[4], s = 0.f;
  #pragma unroll
  for (int i = 0; i < 4; ++i) { v[i] = xr[t + i * 256]; s += v[i]; }
  red[t] = s; __syncthreads();
  for (int st = 128; st > 0; st >>= 1) { if (t < st) red[t] += red[t + st]; __syncthreads(); }
  float mu = red[0] * (1.f / D);
  __syncthreads();
  s = 0.f;
  #pragma unroll
  for (int i = 0; i < 4; ++i) { float d = v[i] - mu; s += d * d; }
  red[t] = s; __syncthreads();
  for (int st = 128; st > 0; st >>= 1) { if (t < st) red[t] += red[t + st]; __syncthreads(); }
  float rstd = rsqrtf(red[0] * (1.f / D) + 1e-5f);
  #pragma unroll
  for (int i = 0; i < 4; ++i) {
    int c = t + i * 256;
    float y = (v[i] - mu) * rstd * gam[c] + bet[c];
    Yf[(size_t)row * D + c] = y;
    if (WB16) Yb[(size_t)row * D + c] = f2bf(y);
  }
}

// ---------------------------------------------------------------- launch
extern "C" void kernel_launch(void* const* d_in, const int* in_sizes, int n_in,
                              void* d_out, int out_size, void* d_ws, size_t ws_size,
                              hipStream_t stream) {
  (void)in_sizes; (void)n_in; (void)out_size; (void)ws_size;
  const float* x     = (const float*)d_in[0];
  const float* w_qkv = (const float*)d_in[1];
  const float* b_qkv = (const float*)d_in[2];
  const float* w_out = (const float*)d_in[3];
  const float* b_out = (const float*)d_in[4];
  const float* w1    = (const float*)d_in[5];
  const float* b1    = (const float*)d_in[6];
  const float* w2    = (const float*)d_in[7];
  const float* b2    = (const float*)d_in[8];
  const float* ln1_g = (const float*)d_in[9];
  const float* ln1_b = (const float*)d_in[10];
  const float* ln2_g = (const float*)d_in[11];
  const float* ln2_b = (const float*)d_in[12];

  const int Bb = 4, T = 2048, D = 1024, DFF = 4096, H = 16;
  const int M = Bb * T;                              // 8192 rows

  char* p = (char*)d_ws;
  auto alloc = [&](size_t bytes) {
    char* r = p; p += (bytes + 255) & ~(size_t)255; return r;
  };
  unsigned short* xb     = (unsigned short*)alloc((size_t)M * D * 2);
  unsigned short* wqkvT  = (unsigned short*)alloc((size_t)D * 3 * D * 2);  // [3D][D]
  unsigned short* woutT  = (unsigned short*)alloc((size_t)D * D * 2);      // [D][D]
  unsigned short* w1T    = (unsigned short*)alloc((size_t)D * DFF * 2);    // [DFF][D]
  unsigned short* w2T    = (unsigned short*)alloc((size_t)DFF * D * 2);    // [D][DFF]
  unsigned short* qkv_b  = (unsigned short*)alloc((size_t)M * 3 * D * 2);
  unsigned short* qkvh   = (unsigned short*)alloc((size_t)3 * M * D * 2);
  unsigned short* attn_b = (unsigned short*)alloc((size_t)M * D * 2);
  float*          y1     = (float*)alloc((size_t)M * D * 4);
  float*          hf     = (float*)alloc((size_t)M * D * 4);
  unsigned short* hb     = (unsigned short*)alloc((size_t)M * D * 2);
  unsigned short* ffb    = (unsigned short*)alloc((size_t)M * DFF * 2);
  float*          y2     = (float*)alloc((size_t)M * D * 4);

  // activations: straight convert; weights: transpose+convert ([K][N]->[N][K])
  k_cvt<<<(M * D + 255) / 256, 256, 0, stream>>>(x, xb, M * D);
  k_cvt_t<<<dim3(3 * D / 32, D / 32),   256, 0, stream>>>(w_qkv, wqkvT, D,   3 * D);
  k_cvt_t<<<dim3(D / 32,     D / 32),   256, 0, stream>>>(w_out, woutT, D,   D);
  k_cvt_t<<<dim3(DFF / 32,   D / 32),   256, 0, stream>>>(w1,    w1T,   D,   DFF);
  k_cvt_t<<<dim3(D / 32,     DFF / 32), 256, 0, stream>>>(w2,    w2T,   DFF, D);

  // QKV projection -> bf16 [M, 3D]
  k_gemm<false, false, false, true><<<dim3(3 * D / 128, M / 128), 256, 0, stream>>>(
      xb, wqkvT, b_qkv, nullptr, nullptr, qkv_b, M, 3 * D, D);
  // repack to [3,B,H,T,64]
  k_repack<<<(M * 3 * D) / 256, 256, 0, stream>>>(qkv_b, qkvh);
  // flash attention
  const size_t hs = (size_t)M * D;
  k_attn<<<dim3(T / 16, H, Bb), 32, 0, stream>>>(qkvh, qkvh + hs, qkvh + 2 * hs, attn_b);
  // out projection + residual(x) -> f32
  k_gemm<false, true, true, false><<<dim3(D / 128, M / 128), 256, 0, stream>>>(
      attn_b, woutT, b_out, x, y1, nullptr, M, D, D);
  // LN1 -> f32 (for residual2) + bf16 (FFN A)
  k_ln<true><<<M, 256, 0, stream>>>(y1, ln1_g, ln1_b, hf, hb);
  // FFN1: ReLU -> bf16
  k_gemm<true, false, false, true><<<dim3(DFF / 128, M / 128), 256, 0, stream>>>(
      hb, w1T, b1, nullptr, nullptr, ffb, M, DFF, D);
  // FFN2 + residual(h) -> f32
  k_gemm<false, true, true, false><<<dim3(D / 128, M / 128), 256, 0, stream>>>(
      ffb, w2T, b2, hf, y2, nullptr, M, D, DFF);
  // LN2 -> final output f32
  k_ln<false><<<M, 256, 0, stream>>>(y2, ln2_g, ln2_b, (float*)d_out, nullptr);
}
